// GCN_4329327034521
// MI455X (gfx1250) — matrix operand order repaired and checked by reading, not compiled, not run on hardware
//
#include <hip/hip_runtime.h>
#include <hip/hip_bf16.h>

typedef __bf16 v16bf __attribute__((ext_vector_type(16)));
typedef __bf16 v4bf  __attribute__((ext_vector_type(4)));
typedef float  v8f   __attribute__((ext_vector_type(8)));

#define HD 128  // hidden/feature dim (F_IN == H == 128)

// ---------------- utility ----------------
__global__ void GCN_zero_f32(float* p, int n) {
    int i = blockIdx.x * blockDim.x + threadIdx.x;
    if (i < n) p[i] = 0.0f;
}

// ---------------- degree ----------------
__global__ void GCN_deg_init(float* deg, int n) {
    int i = blockIdx.x * blockDim.x + threadIdx.x;
    if (i < n) deg[i] = 1.0f;   // self-loop
}
__global__ void GCN_deg_count(const long long* __restrict__ ei, float* deg, int nE) {
    int e = blockIdx.x * blockDim.x + threadIdx.x;
    if (e < nE) atomicAdd(&deg[(int)ei[(long long)nE + e]], 1.0f);
}
__global__ void GCN_deg_finalize(float* deg, int n) {
    int i = blockIdx.x * blockDim.x + threadIdx.x;
    if (i < n) deg[i] = rsqrtf(deg[i]);   // deg -> dinv in place
}

// ---------------- GEMM: Y[rows x 128] = bf16(X) @ bf16(W), WMMA ----------------
// Block = 256 threads = 8 waves; block computes a 128-row x 128-col tile.
// A staged K-swizzled so each lane-half's 16 elements are one contiguous 32B
// LDS read; W staged transposed for the same property on B fragments.
__global__ __launch_bounds__(256) void GCN_gemm_wmma(
    const float* __restrict__ X, const float* __restrict__ W,
    float* __restrict__ Y, int rows) {
    __shared__ __align__(32) __bf16 As[HD * HD];   // 32 KB, K-swizzled rows
    __shared__ __align__(32) __bf16 Wt[HD * HD];   // 32 KB, column-major W
    const int tid  = threadIdx.x;
    const int row0 = blockIdx.x * HD;

    // Stage W transposed: each thread packs 4 consecutive K for one column.
    for (int i = tid; i < HD * HD / 4; i += 256) {
        int col = i >> 5;
        int kb  = (i & 31) * 4;
        v4bf w;
        #pragma unroll
        for (int j = 0; j < 4; ++j) w[j] = (__bf16)W[(size_t)(kb + j) * HD + col];
        *reinterpret_cast<v4bf*>(Wt + col * HD + kb) = w;
    }
    // Stage A: float4 global read, pack to bf16, within each 32-chunk
    // chunk order is [K0..7, K8..15, K16..23, K24..31] (contiguous, matches B).
    for (int i = tid; i < HD * HD / 4; i += 256) {
        int r  = i >> 5;
        int kb = (i & 31) * 4;          // 4-aligned K
        int gr = row0 + r;
        float4 v = {0.f, 0.f, 0.f, 0.f};
        if (gr < rows) v = *reinterpret_cast<const float4*>(X + (size_t)gr * HD + kb);
        int b    = (kb >> 3) & 3;
        int slot = b;                              // identity: 0->0, 1->1, 2->2, 3->3
        int pos  = (kb & ~31) | (slot << 3) | (kb & 7);
        v4bf p;
        p[0] = (__bf16)v.x; p[1] = (__bf16)v.y; p[2] = (__bf16)v.z; p[3] = (__bf16)v.w;
        *reinterpret_cast<v4bf*>(As + r * HD + pos) = p;
    }
    __syncthreads();

    const int lane = tid & 31;
    const int wv   = tid >> 5;        // wave id 0..7
    const int lr   = lane & 15;
    const int hi   = lane >> 4;       // lane half

    v8f acc[8] = {};
    const int arow = wv * 16 + lr;

    for (int k0 = 0; k0 < HD; k0 += 32) {
        // A 16x32 fragment: one contiguous 32B read thanks to the K-swizzle.
        v16bf a = *reinterpret_cast<const v16bf*>(As + arow * HD + k0 + hi * 16);
        #pragma unroll
        for (int n = 0; n < 8; ++n) {
            // B 32x16 fragment: col = n*16+lr, K = k0 + hi*16 + 0..15 contiguous.
            v16bf b = *reinterpret_cast<const v16bf*>(
                Wt + (n * 16 + lr) * HD + k0 + hi * 16);
            acc[n] = __builtin_amdgcn_wmma_f32_16x16x32_bf16(
                false, a, false, b, (short)0, acc[n], false, false);
        }
    }

    // C/D layout: VGPR r holds M = r (lanes 0-15) / r+8 (lanes 16-31), N = lr.
    const int mrow = row0 + wv * 16 + hi * 8;
    if (row0 + HD <= rows) {
        // Fast path: no bounds checks; stores use immediate offsets off one base.
        #pragma unroll
        for (int n = 0; n < 8; ++n) {
            float* yp = Y + (size_t)mrow * HD + n * 16 + lr;
            #pragma unroll
            for (int r = 0; r < 8; ++r)
                yp[r * HD] = acc[n][r];
        }
    } else {
        #pragma unroll
        for (int n = 0; n < 8; ++n) {
            int col = n * 16 + lr;
            #pragma unroll
            for (int r = 0; r < 8; ++r)
                if (mrow + r < rows) Y[(size_t)(mrow + r) * HD + col] = acc[n][r];
        }
    }
}

// ---------------- scatter: init with bias + self-loop term ----------------
__global__ void GCN_scatter_init(const float* __restrict__ hw,
                                 const float* __restrict__ dinv,
                                 const float* __restrict__ bias,
                                 float* __restrict__ agg, int rows) {
    int idx = blockIdx.x * blockDim.x + threadIdx.x;
    if (idx >= rows * HD) return;
    int i = idx >> 7, c = idx & 127;
    float di = dinv[i];
    agg[idx] = bias[c] + hw[idx] * di * di;
}

// One wave per edge: 32 lanes x float4 = 128 features; 4 f32 atomics per lane.
__global__ __launch_bounds__(256) void GCN_scatter_edges(
    const float* __restrict__ hw, const float* __restrict__ dinv,
    const long long* __restrict__ ei, float* __restrict__ agg, int nE) {
    int e = blockIdx.x * 8 + (threadIdx.x >> 5);
    if (e >= nE) return;
    int lane = threadIdx.x & 31;
    int s = (int)ei[e];
    int d = (int)ei[(long long)nE + e];
    float coef = dinv[s] * dinv[d];
    float4 v = ((const float4*)(hw + (size_t)s * HD))[lane];
    float* od = agg + (size_t)d * HD + lane * 4;
    atomicAdd(od + 0, v.x * coef);
    atomicAdd(od + 1, v.y * coef);
    atomicAdd(od + 2, v.z * coef);
    atomicAdd(od + 3, v.w * coef);
}

// ---------------- BatchNorm (training stats) + ReLU ----------------
__global__ __launch_bounds__(128) void GCN_bn_stats(
    const float* __restrict__ x, float* __restrict__ stats, int rows) {
    int c = threadIdx.x;                 // one column per thread
    float s = 0.0f, sq = 0.0f;
    for (int r = blockIdx.x; r < rows; r += gridDim.x) {
        __builtin_prefetch(&x[(size_t)(r + gridDim.x) * HD + c], 0, 0);
        float v = x[(size_t)r * HD + c];
        s += v; sq += v * v;
    }
    atomicAdd(&stats[c], s);
    atomicAdd(&stats[HD + c], sq);
}

__global__ void GCN_bn_apply_relu(float* __restrict__ x,
                                  const float* __restrict__ stats,
                                  const float* __restrict__ g,
                                  const float* __restrict__ be,
                                  int rows) {
    int idx = blockIdx.x * blockDim.x + threadIdx.x;
    if (idx >= rows * HD) return;
    int c = idx & 127;
    float invN = 1.0f / (float)rows;
    float m = stats[c] * invN;
    float v = stats[HD + c] * invN - m * m;
    float xn = (x[idx] - m) * rsqrtf(v + 1e-5f);
    float y  = xn * g[c] + be[c];
    x[idx] = fmaxf(y, 0.0f);
}

// ---------------- global mean pool + final linear ----------------
__global__ __launch_bounds__(256) void GCN_pool_accum(
    const float* __restrict__ h, const long long* __restrict__ batch,
    float* __restrict__ pooled, float* __restrict__ cnt, int rows) {
    int i = blockIdx.x * 8 + (threadIdx.x >> 5);
    if (i >= rows) return;
    int lane = threadIdx.x & 31;
    int b = (int)batch[i];
    float4 v = ((const float4*)(h + (size_t)i * HD))[lane];
    float* p = pooled + (size_t)b * HD + lane * 4;
    atomicAdd(p + 0, v.x);
    atomicAdd(p + 1, v.y);
    atomicAdd(p + 2, v.z);
    atomicAdd(p + 3, v.w);
    if (lane == 0) atomicAdd(&cnt[b], 1.0f);
}

__global__ __launch_bounds__(256) void GCN_final(
    const float* __restrict__ pooled, const float* __restrict__ cnt,
    const float* __restrict__ Wl, const float* __restrict__ bl,
    float* __restrict__ out, int nG) {
    int g = blockIdx.x * 8 + (threadIdx.x >> 5);
    if (g >= nG) return;
    int lane = threadIdx.x & 31;
    float acc = 0.0f;
    #pragma unroll
    for (int j = lane; j < HD; j += 32)
        acc += pooled[(size_t)g * HD + j] * Wl[j];
    #pragma unroll
    for (int off = 16; off > 0; off >>= 1)
        acc += __shfl_xor(acc, off, 32);
    if (lane == 0) out[g] = acc / fmaxf(cnt[g], 1.0f) + bl[0];
}

// ---------------- host ----------------
extern "C" void kernel_launch(void* const* d_in, const int* in_sizes, int n_in,
                              void* d_out, int out_size, void* d_ws, size_t ws_size,
                              hipStream_t stream) {
    const float*     x     = (const float*)d_in[0];
    const long long* ei    = (const long long*)d_in[1];
    const long long* batch = (const long long*)d_in[2];
    const float* Wm[3] = {(const float*)d_in[3],  (const float*)d_in[7],  (const float*)d_in[11]};
    const float* bm[3] = {(const float*)d_in[4],  (const float*)d_in[8],  (const float*)d_in[12]};
    const float* gm[3] = {(const float*)d_in[5],  (const float*)d_in[9],  (const float*)d_in[13]};
    const float* em[3] = {(const float*)d_in[6],  (const float*)d_in[10], (const float*)d_in[14]};
    const float* Wl = (const float*)d_in[15];
    const float* bl = (const float*)d_in[16];
    float* out = (float*)d_out;

    const int N = in_sizes[0] / HD;
    const int E = in_sizes[1] / 2;
    const int G = out_size;

    float* ws     = (float*)d_ws;
    float* hw     = ws;                          // N*128
    float* agg    = hw  + (size_t)N * HD;        // N*128
    float* dinv   = agg + (size_t)N * HD;        // N
    float* stats  = dinv + N;                    // 256
    float* pooled = stats + 2 * HD;              // G*128
    float* cnt    = pooled + (size_t)G * HD;     // G  (contiguous with pooled)

    // normalized-degree coefficients
    GCN_deg_init    <<<(N + 255) / 256, 256, 0, stream>>>(dinv, N);
    GCN_deg_count   <<<(E + 255) / 256, 256, 0, stream>>>(ei, dinv, E);
    GCN_deg_finalize<<<(N + 255) / 256, 256, 0, stream>>>(dinv, N);

    const int elemBlocks = (N * HD + 255) / 256;
    const float* in = x;
    for (int l = 0; l < 3; ++l) {
        GCN_gemm_wmma   <<<(N + HD - 1) / HD, 256, 0, stream>>>(in, Wm[l], hw, N);
        GCN_scatter_init<<<elemBlocks, 256, 0, stream>>>(hw, dinv, bm[l], agg, N);
        GCN_scatter_edges<<<(E + 7) / 8, 256, 0, stream>>>(hw, dinv, ei, agg, E);
        GCN_zero_f32    <<<1, 256, 0, stream>>>(stats, 2 * HD);
        GCN_bn_stats    <<<512, 128, 0, stream>>>(agg, stats, N);
        GCN_bn_apply_relu<<<elemBlocks, 256, 0, stream>>>(agg, stats, gm[l], em[l], N);
        in = agg;   // next layer reads BN output; gemm writes hw (disjoint)
    }

    GCN_zero_f32  <<<(G * HD + G + 255) / 256, 256, 0, stream>>>(pooled, G * HD + G);
    GCN_pool_accum<<<(N + 7) / 8, 256, 0, stream>>>(agg, batch, pooled, cnt, N);
    GCN_final     <<<(G + 7) / 8, 256, 0, stream>>>(pooled, cnt, Wl, bl, out, G);
}
